// NEQUIPLayerFlax_4750233830170
// MI455X (gfx1250) — matrix operand order repaired and checked by reading, not compiled, not run on hardware
//
#include <hip/hip_runtime.h>
#include <math.h>

// ---------------------------------------------------------------------------
// NequIP-style equivariant GNN layer for MI455X (gfx1250, wave32, WMMA).
// Dense contractions ride v_wmma_f32_16x16x32_f16 with f32 accumulate.
// WMMA operands (all but the W_down_s B gather) are staged in LDS so fragments load as 128-bit DS ops:
//   A: row-major tile   -> two ds_load_b128 per fragment
//   B: transposed tile  -> two ds_load_b128 per fragment
// CG tensor products are fused VALU; scatter is hardware f32 atomics into an
// L2-resident accumulator; W2 column blocks are prefetched (global_prefetch).
// ---------------------------------------------------------------------------

typedef __attribute__((ext_vector_type(16))) _Float16 v16h;
typedef __attribute__((ext_vector_type(8)))  _Float16 v8h;
typedef __attribute__((ext_vector_type(8)))  float    v8f;

#define N_NODES 4096
#define N_EDGES 131072

// B-fragment K index for element e (used for residual global gathers).
__device__ __forceinline__ int bk_of(int lane, int e) {
  return e + ((lane & 16) ? 16 : 0);
}

// A fragment (16x32 f16): lane holds row M=lane%16; its 16 halves are
// K = base..base+15 contiguous, base = (lane&16)?16:0.
// p points at &tile[row*ld + ks*32].
__device__ __forceinline__ v16h load_a_frag(const _Float16* p, int lane) {
  union { v16h v; v8h h[2]; } u;
  const int base = (lane & 16) ? 16 : 0;
  u.h[0] = *(const v8h*)(p + base);
  u.h[1] = *(const v8h*)(p + base + 8);
  return u.v;
}

// B fragment (32x16 f16): lane holds col N=lane%16; with the tile stored
// TRANSPOSED (Bt[col*ld + k]) its 16 halves are K = base..base+15 contiguous,
// base = (lane&16)?16:0.  pt points at &Bt[col*ld + ks*32].
__device__ __forceinline__ v16h load_b_frag(const _Float16* pt, int lane) {
  union { v16h v; v8h h[2]; } u;
  const int base = (lane & 16) ? 16 : 0;
  u.h[0] = *(const v8h*)(pt + base);
  u.h[1] = *(const v8h*)(pt + base + 8);
  return u.v;
}

__device__ __forceinline__ float swishf(float x) { return x / (1.f + __expf(-x)); }

__device__ __forceinline__ void atomAddF(float* p, float v) {
  __hip_atomic_fetch_add(p, v, __ATOMIC_RELAXED, __HIP_MEMORY_SCOPE_AGENT);
}

__device__ __forceinline__ v8f wmma16(v16h a, v16h b, v8f c) {
  return __builtin_amdgcn_wmma_f32_16x16x32_f16(false, a, false, b, (short)0, c,
                                                false, false);
}

// ---------------------------------------------------------------------------
__global__ __launch_bounds__(256) void k_zero(float* __restrict__ p, int n) {
  int i = blockIdx.x * blockDim.x + threadIdx.x;
  if (i < n) p[i] = 0.f;
}

// ---------------------------------------------------------------------------
// Kernel 1: up-projections.  blockIdx.y = slice (0:s, 1..3:v_d, 4..8:t_d).
// 64 nodes per block, (64x64)@(64x64) GEMM via 16 WMMA tiles (8 waves x2).
// ---------------------------------------------------------------------------
__global__ __launch_bounds__(256) void k_node_up(
    const float* __restrict__ nf, const float* __restrict__ Wus,
    const float* __restrict__ Wuv, const float* __restrict__ Wut,
    float* __restrict__ s_up, float* __restrict__ v_up,
    float* __restrict__ t_up) {
  __shared__ __align__(32) _Float16 Ah[64 * 64];
  __shared__ __align__(32) _Float16 Bht[64 * 64];  // transposed [col][k]
  const int slice = blockIdx.y;
  const int n0 = blockIdx.x * 64;
  const int tid = threadIdx.x, wave = tid >> 5, lane = tid & 31;

  int off, stride;
  const float* W;
  if (slice == 0)      { off = 0;                 stride = 1; W = Wus; }
  else if (slice < 4)  { off = 64 + (slice - 1);  stride = 3; W = Wuv; }
  else                 { off = 256 + (slice - 4); stride = 5; W = Wut; }

  for (int i = tid; i < 64 * 64; i += 256) {
    int r = i >> 6, cc = i & 63;
    Ah[i] = (_Float16)nf[(n0 + r) * 576 + off + cc * stride];
    Bht[cc * 64 + r] = (_Float16)W[r * 64 + cc];  // r=k, cc=col (coalesced read)
  }
  __syncthreads();

  #pragma unroll
  for (int jj = 0; jj < 2; ++jj) {
    int job = wave * 2 + jj, rt = job >> 2, ct = job & 3;
    int arow = rt * 16 + (lane & 15);
    int bcol = ct * 16 + (lane & 15);
    v8f acc = {};
    #pragma unroll
    for (int ks = 0; ks < 2; ++ks) {
      v16h a = load_a_frag(&Ah[arow * 64 + ks * 32], lane);
      v16h b = load_b_frag(&Bht[bcol * 64 + ks * 32], lane);
      acc = wmma16(a, b, acc);
    }
    #pragma unroll
    for (int e = 0; e < 8; ++e) {
      int row = rt * 16 + e + ((lane & 16) ? 8 : 0);
      int node = n0 + row;
      float v = acc[e];
      if (slice == 0)      s_up[node * 64 + bcol] = v;
      else if (slice < 4)  v_up[node * 192 + (slice - 1) * 64 + bcol] = v;
      else                 t_up[node * 320 + (slice - 4) * 64 + bcol] = v;
    }
  }
}

// ---------------------------------------------------------------------------
// Kernel 2: fused edge kernel.  64 edges/block, 8 waves.
//  Phase A: geometry (n-hat, sh2, radial basis) + transposed W0/W1 -> LDS.
//  Phase B: MLP 8->64->64 via WMMA (swish fused), then stream the 64x1024
//           third layer 64 columns (=4 channels) at a time: coalesced stage of
//           the W2 block into LDS (transposed f16), WMMA, then consume the
//           w-tile immediately in the CG tensor product + atomic scatter.
// ---------------------------------------------------------------------------
__global__ __launch_bounds__(256) void k_edge(
    const float* __restrict__ vectors, const int* __restrict__ senders,
    const int* __restrict__ receivers, const float* __restrict__ W0,
    const float* __restrict__ W1, const float* __restrict__ W2,
    const float* __restrict__ s_up, const float* __restrict__ v_up,
    const float* __restrict__ t_up, float* __restrict__ a0,
    float* __restrict__ a1, float* __restrict__ a2, float* __restrict__ a3) {
  __shared__ float edat[64][8];                    // nx,ny,nz, sh2[0..4]
  __shared__ __align__(32) _Float16 hL[64][32];    // radial, K-padded to 32
  __shared__ __align__(32) _Float16 W0t[64 * 32];  // transposed, K-padded
  __shared__ __align__(32) _Float16 W1t[64 * 64];  // transposed
  __shared__ __align__(32) _Float16 W2t[64 * 64];  // transposed column block
  __shared__ __align__(32) _Float16 act1[64 * 64];
  __shared__ __align__(32) _Float16 act2[64 * 64];
  __shared__ float wL[64 * 64];                    // 4 channels x 16 paths
  __shared__ int esnd[64];
  __shared__ int ercv[64];

  const int tid = threadIdx.x, wave = tid >> 5, lane = tid & 31;
  const int e0 = blockIdx.x * 64;

  // transposed weight staging (coalesced global reads)
  for (int i = tid; i < 2048; i += 256) {
    int k = i >> 6, col = i & 63;
    W0t[col * 32 + k] = (_Float16)((k < 8) ? W0[k * 64 + col] : 0.f);
  }
  for (int i = tid; i < 4096; i += 256) {
    int k = i >> 6, col = i & 63;
    W1t[col * 64 + k] = (_Float16)W1[k * 64 + col];
  }

  if (tid < 64) {
    const int e = e0 + tid;
    float vx = vectors[e * 3 + 0], vy = vectors[e * 3 + 1], vz = vectors[e * 3 + 2];
    float L = sqrtf(vx * vx + vy * vy + vz * vz);
    float invL = (L > 0.f) ? (1.f / L) : 0.f;
    float nx = vx * invL, ny = vy * invL, nz = vz * invL;
    edat[tid][0] = nx; edat[tid][1] = ny; edat[tid][2] = nz;
    edat[tid][3] = 3.872983346207417f  * nx * ny;
    edat[tid][4] = 3.872983346207417f  * ny * nz;
    edat[tid][5] = 1.1180339887498949f * (3.f * nz * nz - 1.f);
    edat[tid][6] = 3.872983346207417f  * nx * nz;
    edat[tid][7] = 1.9364916731037085f * (nx * nx - ny * ny);
    float x = L;
    float xs = (x == 0.f) ? 1.f : x;
    float x2 = x * x, x3 = x2 * x, x6 = x3 * x3, x7 = x6 * x, x8 = x7 * x;
    float env = (x < 1.f) ? (1.f - 28.f * x6 + 48.f * x7 - 21.f * x8) : 0.f;
    float sc = 1.4142135623730951f * env / xs;
    #pragma unroll
    for (int n = 1; n <= 8; ++n)
      hL[tid][n - 1] = (_Float16)(sc * sinf(3.14159265358979f * (float)n * x));
    #pragma unroll
    for (int k = 8; k < 32; ++k) hL[tid][k] = (_Float16)0.f;
    esnd[tid] = senders[e];
    ercv[tid] = receivers[e];
  }
  __syncthreads();

  // ---- MLP layer 1: (64x8 zero-padded to K=32) @ W0(8x64), swish ----
  #pragma unroll
  for (int jj = 0; jj < 2; ++jj) {
    int job = wave * 2 + jj, rt = job >> 2, ct = job & 3;
    int arow = rt * 16 + (lane & 15);
    int bcol = ct * 16 + (lane & 15);
    v16h a = load_a_frag(&hL[arow][0], lane);
    v16h b = load_b_frag(&W0t[bcol * 32], lane);
    v8f acc = {};
    acc = wmma16(a, b, acc);
    #pragma unroll
    for (int e = 0; e < 8; ++e) {
      int row = rt * 16 + e + ((lane & 16) ? 8 : 0);
      act1[row * 64 + bcol] = (_Float16)swishf(acc[e]);
    }
  }
  __syncthreads();

  // ---- MLP layer 2: act1 @ W1(64x64), swish ----
  #pragma unroll
  for (int jj = 0; jj < 2; ++jj) {
    int job = wave * 2 + jj, rt = job >> 2, ct = job & 3;
    int arow = rt * 16 + (lane & 15);
    int bcol = ct * 16 + (lane & 15);
    v8f acc = {};
    #pragma unroll
    for (int ks = 0; ks < 2; ++ks) {
      v16h a = load_a_frag(&act1[arow * 64 + ks * 32], lane);
      v16h b = load_b_frag(&W1t[bcol * 64 + ks * 32], lane);
      acc = wmma16(a, b, acc);
    }
    #pragma unroll
    for (int e = 0; e < 8; ++e) {
      int row = rt * 16 + e + ((lane & 16) ? 8 : 0);
      act2[row * 64 + bcol] = (_Float16)swishf(acc[e]);
    }
  }
  __syncthreads();

  // ---- MLP layer 3 streamed + tensor product ----
  const int rt3 = wave & 3;      // row tile owned by this wave
  const int chalf = wave >> 2;   // which 32-column half
  v16h afA = load_a_frag(&act2[(rt3 * 16 + (lane & 15)) * 64], lane);
  v16h afB = load_a_frag(&act2[(rt3 * 16 + (lane & 15)) * 64 + 32], lane);

  const float IS3  = 0.5773502691896258f;   // 1/sqrt3
  const float I2S3 = 0.2886751345948129f;   // 1/(2*sqrt3)
  const float I5   = 0.4472135954999579f;   // 1/sqrt5
  const float C121 = 0.6324555320336759f;   // sqrt(2/5)
  const float C211 = 1.0954451150103321f;   // sqrt(6/5)
  const float C112 = 1.0954451150103321f;   // sqrt(6/5)
  const float S5   = 2.23606797749979f;     // sqrt5
  const float T5   = 0.8944271909999159f;   // 2/sqrt5
  const float G222 = 1.6561570f;            // sqrt(96/35)
  const float INV3 = 0.5773502691896258f;
  const float INV5 = 0.4472135954999579f;

  for (int cg = 0; cg < 16; ++cg) {
    // prefetch next W2 column block into GL2 (global_prefetch_b8)
    if (cg < 15 && tid < 64)
      __builtin_prefetch(&W2[tid * 1024 + (cg + 1) * 64], 0, 3);

    // coalesced stage of W2 block [64k x 64col], transposed to f16
    for (int i = tid; i < 4096; i += 256) {
      int k = i >> 6, col = i & 63;
      W2t[col * 64 + k] = (_Float16)W2[k * 1024 + cg * 64 + col];
    }
    __syncthreads();

    // w-tile: act2(64x64) @ W2 columns [cg*64, cg*64+64)
    #pragma unroll
    for (int jj = 0; jj < 2; ++jj) {
      int col_local = chalf * 32 + jj * 16 + (lane & 15);
      v8f acc = {};
      v16h b = load_b_frag(&W2t[col_local * 64], lane);
      acc = wmma16(afA, b, acc);
      b = load_b_frag(&W2t[col_local * 64 + 32], lane);
      acc = wmma16(afB, b, acc);
      #pragma unroll
      for (int e = 0; e < 8; ++e) {
        int row = rt3 * 16 + e + ((lane & 16) ? 8 : 0);
        wL[row * 64 + col_local] = acc[e];
      }
    }
    __syncthreads();

    // tensor product: one (edge, channel) per thread
    {
      const int el = tid & 63;
      const int cl = tid >> 6;
      const int c = cg * 4 + cl;
      const int snd = esnd[el];
      const int rcv = ercv[el];
      const float ms  = s_up[snd * 64 + c];
      const float mv0 = v_up[snd * 192 + c];          // l-order: y
      const float mv1 = v_up[snd * 192 + 64 + c];     // z
      const float mv2 = v_up[snd * 192 + 128 + c];    // x
      float mt[5];
      #pragma unroll
      for (int k = 0; k < 5; ++k) mt[k] = t_up[snd * 320 + k * 64 + c];
      const float* wp = &wL[el * 64 + cl * 16];

      float nx = edat[el][0], ny = edat[el][1], nz = edat[el][2];
      float s20 = edat[el][3], s21 = edat[el][4], s22 = edat[el][5];
      float s23 = edat[el][6], s24 = edat[el][7];
      float ux = mv2, uy = mv0, uz = mv1;

      // symmetric traceless matrices from l=2 features / sh2
      float Axx = -mt[2] * I2S3 + 0.5f * mt[4];
      float Ayy = -mt[2] * I2S3 - 0.5f * mt[4];
      float Azz =  mt[2] * IS3;
      float Axy = 0.5f * mt[0], Ayz = 0.5f * mt[1], Axz = 0.5f * mt[3];
      float Bxx = -s22 * I2S3 + 0.5f * s24;
      float Byy = -s22 * I2S3 - 0.5f * s24;
      float Bzz =  s22 * IS3;
      float Bxy = 0.5f * s20, Byz = 0.5f * s21, Bxz = 0.5f * s23;

      float udn = ux * nx + uy * ny + uz * nz;
      float Anx = Axx * nx + Axy * ny + Axz * nz;
      float Any = Axy * nx + Ayy * ny + Ayz * nz;
      float Anz = Axz * nx + Ayz * ny + Azz * nz;
      float nAn = nx * Anx + ny * Any + nz * Anz;
      float Bux = Bxx * ux + Bxy * uy + Bxz * uz;
      float Buy = Bxy * ux + Byy * uy + Byz * uz;
      float Buz = Bxz * ux + Byz * uy + Bzz * uz;

      float p220 = (mt[0] * s20 + mt[1] * s21 + mt[2] * s22 + mt[3] * s23 +
                    mt[4] * s24) * I5;
      float out0 = INV3 * (wp[0]  * ms + wp[1]  * udn + wp[2]  * p220);
      float out3 = INV3 * (wp[13] * ms + wp[14] * udn + wp[15] * p220);

      // l=1 output (cartesian, then back to (y,z,x) order)
      float vx1 = wp[3] * ms * nx + wp[4] * ux * IS3 + wp[5] * C121 * Bux +
                  wp[6] * C211 * Anx + wp[7] * (S5 * nAn * nx - T5 * Anx);
      float vy1 = wp[3] * ms * ny + wp[4] * uy * IS3 + wp[5] * C121 * Buy +
                  wp[6] * C211 * Any + wp[7] * (S5 * nAn * ny - T5 * Any);
      float vz1 = wp[3] * ms * nz + wp[4] * uz * IS3 + wp[5] * C121 * Buz +
                  wp[6] * C211 * Anz + wp[7] * (S5 * nAn * nz - T5 * Anz);
      vx1 *= INV5; vy1 *= INV5; vz1 *= INV5;

      // l=2 output
      float q0 = 0.5f * (ux * ny + uy * nx);
      float q1 = 0.5f * (uy * nz + uz * ny);
      float q3 = 0.5f * (ux * nz + uz * nx);
      float q2 = (2.f * uz * nz - ux * nx - uy * ny) * I2S3;
      float q4 = 0.5f * (ux * nx - uy * ny);

      float Xxy = udn * nx * ny - 0.2f * (ux * ny + uy * nx);
      float Xyz = udn * ny * nz - 0.2f * (uy * nz + uz * ny);
      float Xxz = udn * nx * nz - 0.2f * (ux * nz + uz * nx);
      float Xxx = udn * nx * nx - 0.2f * (udn + 2.f * ux * nx);
      float Xyy = udn * ny * ny - 0.2f * (udn + 2.f * uy * ny);
      float Xzz = udn * nz * nz - 0.2f * (udn + 2.f * uz * nz);
      float r0 = Xxy, r1 = Xyz, r3 = Xxz;
      float r2 = (2.f * Xzz - Xxx - Xyy) * I2S3;
      float r4 = 0.5f * (Xxx - Xyy);

      float Gxx = Axx * Bxx + Axy * Bxy + Axz * Bxz;
      float Gxy = Axx * Bxy + Axy * Byy + Axz * Byz;
      float Gxz = Axx * Bxz + Axy * Byz + Axz * Bzz;
      float Gyx = Axy * Bxx + Ayy * Bxy + Ayz * Bxz;
      float Gyy = Axy * Bxy + Ayy * Byy + Ayz * Byz;
      float Gyz = Axy * Bxz + Ayy * Byz + Ayz * Bzz;
      float Gzx = Axz * Bxx + Ayz * Bxy + Azz * Bxz;
      float Gzy = Axz * Bxy + Ayz * Byy + Azz * Byz;
      float Gzz = Axz * Bxz + Ayz * Byz + Azz * Bzz;
      float t0p = G222 * 0.5f * (Gxy + Gyx);
      float t1p = G222 * 0.5f * (Gyz + Gzy);
      float t2p = G222 * (2.f * Gzz - Gxx - Gyy) * I2S3;
      float t3p = G222 * 0.5f * (Gxz + Gzx);
      float t4p = G222 * 0.5f * (Gxx - Gyy);

      float o2_0 = INV5 * (wp[8] * ms * s20 * I5 + wp[9] * C112 * q0 +
                           wp[10] * S5 * r0 + wp[11] * mt[0] * I5 + wp[12] * t0p);
      float o2_1 = INV5 * (wp[8] * ms * s21 * I5 + wp[9] * C112 * q1 +
                           wp[10] * S5 * r1 + wp[11] * mt[1] * I5 + wp[12] * t1p);
      float o2_2 = INV5 * (wp[8] * ms * s22 * I5 + wp[9] * C112 * q2 +
                           wp[10] * S5 * r2 + wp[11] * mt[2] * I5 + wp[12] * t2p);
      float o2_3 = INV5 * (wp[8] * ms * s23 * I5 + wp[9] * C112 * q3 +
                           wp[10] * S5 * r3 + wp[11] * mt[3] * I5 + wp[12] * t3p);
      float o2_4 = INV5 * (wp[8] * ms * s24 * I5 + wp[9] * C112 * q4 +
                           wp[10] * S5 * r4 + wp[11] * mt[4] * I5 + wp[12] * t4p);

      atomAddF(&a0[rcv * 64 + c], out0);
      atomAddF(&a1[rcv * 192 + c], vy1);          // l-order 0 = y
      atomAddF(&a1[rcv * 192 + 64 + c], vz1);     // 1 = z
      atomAddF(&a1[rcv * 192 + 128 + c], vx1);    // 2 = x
      atomAddF(&a2[rcv * 320 + c], o2_0);
      atomAddF(&a2[rcv * 320 + 64 + c], o2_1);
      atomAddF(&a2[rcv * 320 + 128 + c], o2_2);
      atomAddF(&a2[rcv * 320 + 192 + c], o2_3);
      atomAddF(&a2[rcv * 320 + 256 + c], o2_4);
      atomAddF(&a3[rcv * 64 + c], out3);
    }
    __syncthreads();
  }
}

// ---------------------------------------------------------------------------
// Kernel 3: down-projections (WMMA) + species skip (VALU) + swish gating.
// 32 nodes per block, 8 waves.
// ---------------------------------------------------------------------------
__global__ __launch_bounds__(256) void k_down(
    const float* __restrict__ nf, const int* __restrict__ specie,
    const float* __restrict__ a0, const float* __restrict__ a1,
    const float* __restrict__ a2, const float* __restrict__ a3,
    const float* __restrict__ Wds, const float* __restrict__ Wdv,
    const float* __restrict__ Wdt, const float* __restrict__ Wss,
    const float* __restrict__ Wsv, const float* __restrict__ Wst,
    float* __restrict__ out) {
  __shared__ __align__(32) _Float16 AhS[32 * 128];
  __shared__ float sL[32 * 192];  // s_all, later holds gates in cols>=64
  __shared__ float featS[32 * 64];
  __shared__ __align__(32) _Float16 Ah2[32 * 64];
  __shared__ __align__(32) _Float16 Wt_t[64 * 64];  // transposed W_down_{v,t}
  __shared__ float feat2[32 * 64];
  __shared__ int spec[32];

  const int tid = threadIdx.x, wave = tid >> 5, lane = tid & 31;
  const int n0 = blockIdx.x * 32;
  const float INVNB = 0.17677669529663687f;  // 1/sqrt(32)

  for (int i = tid; i < 32 * 128; i += 256) {
    int r = i >> 7, cc = i & 127;
    float v = (cc < 64) ? a0[(n0 + r) * 64 + cc] : a3[(n0 + r) * 64 + (cc - 64)];
    AhS[i] = (_Float16)(v * INVNB);
  }
  for (int i = tid; i < 32 * 64; i += 256) {
    int r = i >> 6, cc = i & 63;
    featS[i] = nf[(n0 + r) * 576 + cc];
  }
  if (tid < 32) spec[tid] = specie[n0 + tid];
  __syncthreads();

  // s_all = [a0|a3](32x128) @ W_down_s(128x192): 2x12 tiles, K=4 steps
  #pragma unroll
  for (int jj = 0; jj < 3; ++jj) {
    int job = wave * 3 + jj;
    int rt = job / 12, ct = job % 12;
    int arow = rt * 16 + (lane & 15);
    int bcol = ct * 16 + (lane & 15);
    v8f acc = {};
    #pragma unroll
    for (int ks = 0; ks < 4; ++ks) {
      v16h a = load_a_frag(&AhS[arow * 128 + ks * 32], lane);
      v16h b;
      #pragma unroll
      for (int e = 0; e < 16; ++e)
        b[e] = (_Float16)Wds[(ks * 32 + bk_of(lane, e)) * 192 + bcol];
      acc = wmma16(a, b, acc);
    }
    #pragma unroll
    for (int e = 0; e < 8; ++e) {
      int row = rt * 16 + e + ((lane & 16) ? 8 : 0);
      sL[row * 192 + bcol] = acc[e];
    }
  }
  __syncthreads();

  // skip_s + swish; cols<64 -> out_s, cols>=64 -> gates (kept in sL)
  for (int i = tid; i < 32 * 192; i += 256) {
    int r = i / 192, col = i % 192;
    const float* Wr = Wss + spec[r] * (64 * 192);
    float s = 0.f;
    #pragma unroll 8
    for (int c2 = 0; c2 < 64; ++c2) s += featS[r * 64 + c2] * Wr[c2 * 192 + col];
    float val = sL[i] + s;
    float sw = swishf(val);
    if (col < 64) out[(n0 + r) * 576 + col] = sw;
    else          sL[i] = sw;
  }
  __syncthreads();

  // v (d=0..2) and t (d=3..7) down-projections + skip + gate
  for (int d = 0; d < 8; ++d) {
    const bool isv = (d < 3);
    const int dd = isv ? d : d - 3;
    const float* asrc = isv ? a1 : a2;
    const int astride = isv ? 192 : 320;
    const float* Wd = isv ? Wdv : Wdt;
    const float* Wsk = isv ? Wsv : Wst;

    for (int i = tid; i < 32 * 64; i += 256) {
      int r = i >> 6, cc = i & 63;
      Ah2[i] = (_Float16)(asrc[(n0 + r) * astride + dd * 64 + cc] * INVNB);
      feat2[i] = isv ? nf[(n0 + r) * 576 + 64 + cc * 3 + dd]
                     : nf[(n0 + r) * 576 + 256 + cc * 5 + dd];
    }
    for (int i = tid; i < 4096; i += 256) {
      int k = i >> 6, col = i & 63;
      Wt_t[col * 64 + k] = (_Float16)Wd[k * 64 + col];
    }
    __syncthreads();

    {
      int rt = wave >> 2, ct = wave & 3;  // 8 jobs, one per wave
      int arow = rt * 16 + (lane & 15);
      int bcol = ct * 16 + (lane & 15);
      v8f acc = {};
      #pragma unroll
      for (int ks = 0; ks < 2; ++ks) {
        v16h a = load_a_frag(&Ah2[arow * 64 + ks * 32], lane);
        v16h b = load_b_frag(&Wt_t[bcol * 64 + ks * 32], lane);
        acc = wmma16(a, b, acc);
      }
      #pragma unroll
      for (int e = 0; e < 8; ++e) {
        int row = rt * 16 + e + ((lane & 16) ? 8 : 0);
        const float* Wr = Wsk + spec[row] * 4096;
        float s = 0.f;
        #pragma unroll 8
        for (int c2 = 0; c2 < 64; ++c2) s += feat2[row * 64 + c2] * Wr[c2 * 64 + bcol];
        float val = acc[e] + s;
        float g = isv ? sL[row * 192 + 64 + bcol] : sL[row * 192 + 128 + bcol];
        float o = val * g;
        int node = n0 + row;
        if (isv) out[node * 576 + 64 + bcol * 3 + dd] = o;
        else     out[node * 576 + 256 + bcol * 5 + dd] = o;
      }
    }
    __syncthreads();
  }
}

// ---------------------------------------------------------------------------
extern "C" void kernel_launch(void* const* d_in, const int* in_sizes, int n_in,
                              void* d_out, int out_size, void* d_ws,
                              size_t ws_size, hipStream_t stream) {
  const float* vectors     = (const float*)d_in[0];
  const float* node_feats  = (const float*)d_in[1];
  const int*   node_specie = (const int*)d_in[2];
  const int*   senders     = (const int*)d_in[3];
  const int*   receivers   = (const int*)d_in[4];
  const float* W_skip_s    = (const float*)d_in[5];
  const float* W_skip_v    = (const float*)d_in[6];
  const float* W_skip_t    = (const float*)d_in[7];
  const float* W_up_s      = (const float*)d_in[8];
  const float* W_up_v      = (const float*)d_in[9];
  const float* W_up_t      = (const float*)d_in[10];
  const float* W_mlp0      = (const float*)d_in[11];
  const float* W_mlp1      = (const float*)d_in[12];
  const float* W_mlp2      = (const float*)d_in[13];
  const float* W_down_s    = (const float*)d_in[14];
  const float* W_down_v    = (const float*)d_in[15];
  const float* W_down_t    = (const float*)d_in[16];
  float* out = (float*)d_out;

  float* ws = (float*)d_ws;
  float* s_up = ws;                      // N*64
  float* v_up = ws + 262144;             // N*192
  float* t_up = ws + 1048576;            // N*320
  float* a0   = ws + 2359296;            // N*64
  float* a1   = ws + 2621440;            // N*192
  float* a2   = ws + 3407872;            // N*320
  float* a3   = ws + 4718592;            // N*64

  const int n_acc = 10 * N_NODES * 64;   // a0..a3 contiguous
  k_zero<<<(n_acc + 255) / 256, 256, 0, stream>>>(a0, n_acc);

  dim3 g1(N_NODES / 64, 9, 1);
  k_node_up<<<g1, 256, 0, stream>>>(node_feats, W_up_s, W_up_v, W_up_t, s_up,
                                    v_up, t_up);

  k_edge<<<N_EDGES / 64, 256, 0, stream>>>(vectors, senders, receivers, W_mlp0,
                                           W_mlp1, W_mlp2, s_up, v_up, t_up, a0,
                                           a1, a2, a3);

  k_down<<<N_NODES / 32, 256, 0, stream>>>(node_feats, node_specie, a0, a1, a2,
                                           a3, W_down_s, W_down_v, W_down_t,
                                           W_skip_s, W_skip_v, W_skip_t, out);
}
